// SelectiveElmanCell_2396591751743
// MI455X (gfx1250) — matrix operand order repaired and checked by reading, not compiled, not run on hardware
//
#include <hip/hip_runtime.h>
#include <hip/hip_bf16.h>
#include <stdint.h>

// ---------------------------------------------------------------------------
// Selective Elman cell for MI455X (gfx1250).
//   Phase 1: pack x and 5 weight matrices to bf16 WMMA tile layouts.
//   Phase 2: fused 4-way projection GEMM (wmma bf16, f32 accum); weight tiles
//            staged block-wide into LDS via global_load_async_to_lds_b128,
//            double-buffered so the async copy for kc+1 overlaps WMMAs for kc.
//   Phase 3: persistent scan kernel, grid barrier per step; the 32 KB packed-h
//            operand is async-staged into LDS once per block per step (all 8
//            waves share it), then 32 WMMAs/wave/step against L2-resident W_h.
// ---------------------------------------------------------------------------

typedef __attribute__((ext_vector_type(16))) __bf16 v16bf;
typedef __attribute__((ext_vector_type(8)))  float  v8f;
typedef int v4i __attribute__((vector_size(16)));

#define DIM    1024
#define TLEN   2048
#define BATCH  8
#define MROWS  (TLEN * BATCH)      // 16384 rows for projection GEMM
#define KTILES (DIM / 32)          // 32 K-tiles of 32
#define NTILES (DIM / 16)          // 64 N-tiles of 16
#define MTILES (MROWS / 16)        // 1024 M-tiles of 16
#define WMAT   (DIM * DIM)         // elements per weight matrix
#define OUTS_ELEMS ((size_t)TLEN * BATCH * DIM)   // 16777216
#define BD     (BATCH * DIM)       // 8192

// ---- workspace layout (bytes) ----
#define OFF_XPACK 0ull
#define SZ_XPACK  ((unsigned long long)MROWS * DIM * 2)   // 32 MB
#define OFF_WPACK (OFF_XPACK + SZ_XPACK)
#define SZ_WPACK  (5ull * WMAT * 2)                       // 10 MB (Wx,WB,WC,Wdt,Wh)
#define OFF_BX    (OFF_WPACK + SZ_WPACK)
#define SZ_GATE   ((unsigned long long)MROWS * DIM * 4)   // 64 MB each
#define OFF_CG    (OFF_BX + SZ_GATE)
#define OFF_DTG   (OFF_CG + SZ_GATE)
#define OFF_HBUF  (OFF_DTG + SZ_GATE)
#define SZ_HBUF   (2ull * 16 * DIM * 2)                   // double-buffered bf16 h
#define OFF_BAR   (OFF_HBUF + SZ_HBUF)

// ---- CDNA5 async global->LDS path (ASYNCcnt family) ----
#if __has_builtin(__builtin_amdgcn_global_load_async_to_lds_b128)
#define USE_ASYNC 1
#else
#define USE_ASYNC 0
#endif

typedef __attribute__((address_space(1))) v4i* gas_v4i;
typedef __attribute__((address_space(3))) v4i* las_v4i;

__device__ __forceinline__ void async_ld16(const void* gptr, void* lptr) {
#if USE_ASYNC
  __builtin_amdgcn_global_load_async_to_lds_b128(
      (gas_v4i)(v4i*)(void*)gptr,
      (las_v4i)(v4i*)lptr, 0, 0);
#else
  *(float4*)lptr = *(const float4*)gptr;   // fallback: stage through VGPRs
#endif
}

__device__ __forceinline__ void wait_async0() {
#if USE_ASYNC
#if __has_builtin(__builtin_amdgcn_s_wait_asynccnt)
  __builtin_amdgcn_s_wait_asynccnt(0);
#else
  asm volatile("s_wait_asynccnt 0x0" ::: "memory");
#endif
#endif
}

// ---- helpers ----
__device__ __forceinline__ __bf16 f2bf(float f) {
  // round-to-nearest-even f32 -> bf16
  unsigned u = __builtin_bit_cast(unsigned, f);
  unsigned r = (u + 0x7FFFu + ((u >> 16) & 1u)) >> 16;
  unsigned short s = (unsigned short)r;
  return __builtin_bit_cast(__bf16, s);
}
__device__ __forceinline__ float sigf(float x) {
  return 1.0f / (1.0f + __expf(-x));
}
__device__ __forceinline__ float tanhf_fast(float x) {
  x = fminf(15.0f, fmaxf(-15.0f, x));
  float e = __expf(2.0f * x);
  return (e - 1.0f) / (e + 1.0f);
}

// A-matrix (16x32 bf16) tile layout: lane L holds row M = L%16, hi = L/16.
// Element e (0..15) of the lane's v16bf: g=e>>3, vv=(e&7)>>1, s=e&1,
//   K = g*16 + hi*8 + 2*vv + s.           (ISA 7.12.2, 16-bit A 16x32)
// B-matrix (32x16 bf16) tile layout: lane L holds col N = L%16, hi = L/16,
//   element e maps to K = (e>>3)*16 + hi*8 + (e&7), same as A.      (ISA 7.12.5 pattern, K=32 dense)

// ---------------------------------------------------------------------------
__global__ void init_kernel(const float* __restrict__ h0,
                            float* __restrict__ hout,     // h output section base
                            __bf16* __restrict__ hb,      // packed h double buffer
                            int* __restrict__ bar) {
  int tid = blockIdx.x * blockDim.x + threadIdx.x;
  int stride = gridDim.x * blockDim.x;
  // h[0] = h0
  for (int i = tid; i < BD; i += stride) hout[i] = h0[i];
  // packed h buffers: buf0 = h0 (rows 8..15 zero), buf1 = zero
  for (int i = tid; i < 2 * 16 * DIM; i += stride) {
    int buf = i / (16 * DIM);
    int r   = i % (16 * DIM);
    int kc  = r >> 9;            // /512
    int rr  = r & 511;
    int lane = rr >> 4;
    int e    = rr & 15;
    int m  = lane & 15;
    int hi = lane >> 4;
    int g  = e >> 3, vv = (e & 7) >> 1, s = e & 1;
    int k  = kc * 32 + g * 16 + hi * 8 + vv * 2 + s;
    float v = (buf == 0 && m < BATCH) ? h0[m * DIM + k] : 0.0f;
    hb[i] = f2bf(v);
  }
  if (tid < 8) bar[tid] = 0;
}

// ---------------------------------------------------------------------------
__global__ void pack_x_kernel(const float* __restrict__ x, __bf16* __restrict__ xp) {
  int tid = blockIdx.x * blockDim.x + threadIdx.x;
  if (tid >= MROWS * DIM) return;
  int tile = tid >> 9;           // (mt*32 + kc)
  int r    = tid & 511;
  int lane = r >> 4;
  int e    = r & 15;
  int mt = tile >> 5, kc = tile & 31;
  int m  = mt * 16 + (lane & 15);
  int hi = lane >> 4;
  int g  = e >> 3, vv = (e & 7) >> 1, s = e & 1;
  int k  = kc * 32 + g * 16 + hi * 8 + vv * 2 + s;
  xp[tid] = f2bf(x[(size_t)m * DIM + k]);
}

__global__ void pack_w_kernel(const float* __restrict__ Wx, const float* __restrict__ WB,
                              const float* __restrict__ WC, const float* __restrict__ Wdt,
                              const float* __restrict__ Wh, __bf16* __restrict__ wp) {
  int tid = blockIdx.x * blockDim.x + threadIdx.x;
  if (tid >= 5 * WMAT) return;
  int mat = tid / WMAT;
  int rem = tid % WMAT;
  int tile = rem >> 9;           // (nt*32 + kc)
  int r    = rem & 511;
  int lane = r >> 4;
  int e    = r & 15;
  int nt = tile >> 5, kc = tile & 31;
  int n  = nt * 16 + (lane & 15);
  int k  = kc * 32 + (e >> 3) * 16 + (lane >> 4) * 8 + (e & 7);
  const float* W = (mat == 0) ? Wx : (mat == 1) ? WB : (mat == 2) ? WC
                  : (mat == 3) ? Wdt : Wh;
  // GEMM computes out = x @ W^T : B-matrix[k][n] = W[n][k] (row of W contiguous in k)
  wp[tid] = f2bf(W[(size_t)n * DIM + k]);
}

// ---------------------------------------------------------------------------
// Projection. Block = 8 waves sharing one nt; waves cover 8 consecutive mt.
// The 4 weight B-tiles per kc step (4 KB) are async-staged into LDS once per
// block (8x traffic reduction on the dominant stream), double-buffered.
__global__ __launch_bounds__(256) void proj_kernel(
    const __bf16* __restrict__ xp, const __bf16* __restrict__ wp,
    const float* __restrict__ bB, const float* __restrict__ bC,
    const float* __restrict__ bdt,
    float* __restrict__ bx, float* __restrict__ cg, float* __restrict__ dtg) {
  __shared__ __align__(32) __bf16 sB[2][4][512];   // [buf][mat][32x16 tile]

  int w    = threadIdx.x >> 5;
  int lane = threadIdx.x & 31;
  int nt = blockIdx.x & 63;
  int mt = (blockIdx.x >> 6) * 8 + w;

  const v16bf* A = (const v16bf*)xp + (size_t)mt * (KTILES * 32) + lane;

  // each wave stages one half-tile of one matrix: mat = w>>1, half = w&1
  int smat = w >> 1, shalf = w & 1;
  const __bf16* gB = wp + (size_t)smat * WMAT + (size_t)nt * (KTILES * 512)
                     + shalf * 256 + lane * 8;
  __bf16* lB0 = &sB[0][smat][shalf * 256 + lane * 8];
  __bf16* lB1 = &sB[1][smat][shalf * 256 + lane * 8];

  async_ld16(gB, lB0);                        // preload kc = 0
  wait_async0();
  __syncthreads();

  v8f a0 = {}, a1 = {}, a2 = {}, a3 = {};
  for (int kc = 0; kc < KTILES; ++kc) {
    int buf = kc & 1;
    if (kc + 1 < KTILES) {
      async_ld16(gB + (kc + 1) * 512, buf ? lB0 : lB1);
      __builtin_prefetch((const void*)(A + (kc + 1) * 32), 0, 1);
    }
    v16bf af = A[kc * 32];
    v16bf b0 = *(const v16bf*)&sB[buf][0][lane * 16];
    v16bf b1 = *(const v16bf*)&sB[buf][1][lane * 16];
    v16bf b2 = *(const v16bf*)&sB[buf][2][lane * 16];
    v16bf b3 = *(const v16bf*)&sB[buf][3][lane * 16];
    a0 = __builtin_amdgcn_wmma_f32_16x16x32_bf16(false, af, false, b0, (short)0, a0, false, false);
    a1 = __builtin_amdgcn_wmma_f32_16x16x32_bf16(false, af, false, b1, (short)0, a1, false, false);
    a2 = __builtin_amdgcn_wmma_f32_16x16x32_bf16(false, af, false, b2, (short)0, a2, false, false);
    a3 = __builtin_amdgcn_wmma_f32_16x16x32_bf16(false, af, false, b3, (short)0, a3, false, false);
    wait_async0();
    __syncthreads();
  }

  int hi = lane >> 4, ln = lane & 15;
  int col = nt * 16 + ln;
  float bBc = bB[col], bCc = bC[col], bdc = bdt[col];
#pragma unroll
  for (int i = 0; i < 8; ++i) {
    int row = mt * 16 + i + 8 * hi;          // C layout: VGPR i, lane -> (M,N)
    size_t idx = (size_t)row * DIM + col;
    float xpv = a0[i];
    float bg  = sigf(a1[i] + bBc);
    float cgv = sigf(a2[i] + bCc);
    float dg  = sigf(a3[i] + bdc);
    bx[idx]  = bg * xpv;                     // fused B_t * xp_t
    cg[idx]  = cgv;
    dtg[idx] = dg;
  }
}

// ---------------------------------------------------------------------------
__device__ __forceinline__ void grid_sync(int* bar, int nblk) {
  __syncthreads();
  if (threadIdx.x == 0) {
    __threadfence();
    volatile int* vgen = bar + 1;
    int g = *vgen;
    if (atomicAdd(bar, 1) == nblk - 1) {
      *bar = 0;
      __threadfence();
      atomicAdd((int*)(bar + 1), 1);
    } else {
      while (*vgen == g) { __builtin_amdgcn_s_sleep(1); }
    }
  }
  __syncthreads();
  __threadfence();
}

// Persistent scan: 8 blocks x 8 waves = 64 waves, one 16-col tile each.
// W_h (2 MB bf16) stays L2-resident. Each step, the 32 KB packed-h operand
// (identical for every wave) is async-staged into LDS once per block.
__global__ __launch_bounds__(256) void scan_kernel(
    const __bf16* __restrict__ whp,    // packed W_h (B-layout tiles)
    __bf16* __restrict__ hb,           // 2 x [16 x DIM] packed A-layout
    const float* __restrict__ bx, const float* __restrict__ cgb,
    const float* __restrict__ dtgb,
    const float* __restrict__ x, const float* __restrict__ bvec,
    const float* __restrict__ bgate,
    float* __restrict__ outs, float* __restrict__ hout, int* bar) {
  __shared__ __align__(32) __bf16 sA[16 * DIM];    // 32 KB packed h (A-layout)

  int wave = blockIdx.x * 8 + (threadIdx.x >> 5);
  int lane = threadIdx.x & 31;
  int nt = wave;                                  // 0..63
  const v16bf* Wt = (const v16bf*)whp + (size_t)nt * (KTILES * 32) + lane;
  int hi = lane >> 4, ln = lane & 15;
  int col = nt * 16 + ln;
  float bv = bvec[col], bg2 = bgate[col];

  // precompute pack coordinates for this lane's column (K = col for next step)
  int kc2 = col >> 5, kk = col & 31;
  int g = kk >> 4, r2 = kk & 15;
  int hi2 = r2 >> 3, rr = r2 & 7, vv = rr >> 1, ss = rr & 1;
  int ee = (g << 3) + (vv << 1) + ss;

  for (int t = 0; t < TLEN; ++t) {
    int cur = t & 1, nxt = cur ^ 1;
    const __bf16* hbase = hb + (size_t)cur * 16 * DIM;
    // stage full packed-h (16384 bf16 = 32 KB) into LDS: 256 thr x 8 x 16 B
#pragma unroll
    for (int j = 0; j < 8; ++j) {
      int off = (j * 256 + threadIdx.x) * 8;      // bf16 elements, 16B chunks
      async_ld16(hbase + off, &sA[off]);
    }
    wait_async0();
    __syncthreads();

    v8f acc = {};
#pragma unroll 4
    for (int kc = 0; kc < KTILES; ++kc) {
      v16bf af  = *(const v16bf*)&sA[kc * 512 + lane * 16];
      v16bf bf_ = Wt[kc * 32];
      acc = __builtin_amdgcn_wmma_f32_16x16x32_bf16(false, af, false, bf_, (short)0, acc, false, false);
    }
    if (hi == 0) {                                // rows 0..7 valid (batch=8)
#pragma unroll
      for (int i = 0; i < 8; ++i) {
        int m = i;
        size_t idx = ((size_t)t * BATCH + m) * DIM + col;
        float raw  = bx[idx] + acc[i] + bv;       // B_t*xp + h@Wh^T + b
        float cand = tanhf_fast(raw);
        float dt   = dtgb[idx];
        float hp   = hout[(size_t)t * BD + m * DIM + col];
        float hn   = (1.0f - dt) * hp + dt * cand;
        float z    = hn + x[idx] + bg2;
        float gate = z * sigf(z);                 // silu
        outs[idx]  = cgb[idx] * hn * gate;
        hout[((size_t)t + 1) * BD + m * DIM + col] = hn;
        hb[(size_t)nxt * 16 * DIM + kc2 * 512 + (m + (hi2 << 4)) * 16 + ee] = f2bf(hn);
      }
    }
    grid_sync(bar, 8);   // also protects sA (staged again next iteration)
  }
}

// ---------------------------------------------------------------------------
extern "C" void kernel_launch(void* const* d_in, const int* in_sizes, int n_in,
                              void* d_out, int out_size, void* d_ws, size_t ws_size,
                              hipStream_t stream) {
  const float* x    = (const float*)d_in[0];
  const float* h0   = (const float*)d_in[1];
  const float* W_x  = (const float*)d_in[2];
  const float* W_h  = (const float*)d_in[3];
  const float* W_B  = (const float*)d_in[4];
  const float* W_C  = (const float*)d_in[5];
  const float* W_dt = (const float*)d_in[6];
  const float* b    = (const float*)d_in[7];
  const float* b_B  = (const float*)d_in[8];
  const float* b_C  = (const float*)d_in[9];
  const float* b_dt = (const float*)d_in[10];
  const float* b_g  = (const float*)d_in[11];

  float* outs = (float*)d_out;
  float* hout = (float*)d_out + OUTS_ELEMS;      // h section: [T+1, B, D]

  uint8_t* ws    = (uint8_t*)d_ws;
  __bf16*  xpack = (__bf16*)(ws + OFF_XPACK);
  __bf16*  wpack = (__bf16*)(ws + OFF_WPACK);
  float*   bx    = (float*)(ws + OFF_BX);
  float*   cg    = (float*)(ws + OFF_CG);
  float*   dtg   = (float*)(ws + OFF_DTG);
  __bf16*  hb    = (__bf16*)(ws + OFF_HBUF);
  int*     bar   = (int*)(ws + OFF_BAR);

  init_kernel<<<64, 256, 0, stream>>>(h0, hout, hb, bar);
  pack_x_kernel<<<(MROWS * DIM + 255) / 256, 256, 0, stream>>>(x, xpack);
  pack_w_kernel<<<(5 * WMAT + 255) / 256, 256, 0, stream>>>(W_x, W_B, W_C, W_dt, W_h, wpack);
  proj_kernel<<<(MTILES * NTILES) / 8, 256, 0, stream>>>(xpack, wpack, b_B, b_C, b_dt,
                                                         bx, cg, dtg);
  scan_kernel<<<8, 256, 0, stream>>>(wpack + (size_t)4 * WMAT, hb, bx, cg, dtg,
                                     x, b, b_g, outs, hout, bar);
}
